// MolTransformer_49452253446998
// MI455X (gfx1250) — compile-verified, matrix operand order checked by reading, not run on hardware
//
#include <hip/hip_runtime.h>
#include <hip/hip_bf16.h>
#include <stdint.h>

// ---------------- problem constants ----------------
#define NB    32      // batch
#define NN    64      // atoms per mol
#define HH    8       // heads
#define DK    80
#define FA    82
#define FP    102
#define HID   256
#define CIN   722     // FA + H*DK
#define CPAD  736     // CIN padded to mult of 32

typedef __attribute__((ext_vector_type(16))) __bf16 v16bf;
typedef __attribute__((ext_vector_type(8)))  float  v8f;

// ---------------- workspace layout (bytes) ----------------
#define OFF_P_ATTN   0ull          // bf16 [32*64*64, 80]
#define OFF_P_MSG    20971520ull   // bf16 [32*64*64, 80]
#define OFF_ATOM_H0  41943040ull   // f32  [256*64, 80]
#define OFF_ATOM_H   47185920ull   // f32  [256*64, 80]
#define OFF_AI       52428800ull   // f32  [256*64, 80]
#define OFF_AJ       57671680ull   // f32  [256*64, 80]
#define OFF_AIM      62914560ull   // f32  [256*64, 80]
#define OFF_AJMT     68157440ull   // bf16 [256][80][64]  (AjM transposed per hb)
#define OFF_W2       73400320ull   // f32  [256*64, 80]
#define OFF_PROBS    78643200ull   // bf16 [256*64, 64]
#define OFF_ROWSUM   82837504ull   // f32  [256*64]
#define OFF_CONCAT   82903040ull   // bf16 [2048, 736]
#define OFF_WTS      85917696ull   // bf16 weights (transposed, padded)

// weight region sub-offsets (ushort units)
#define W_ATOMI_T 0        // [640][96]   W_atom_i^T, K-pad 82->96
#define W_1A      61440    // [80][96]    W_attn_h rows 0:80  ^T
#define W_2A      69120    // [80][96]    W_attn_h rows 80:160 ^T
#define W_1M      76800    // [80][96]    W_msg_h rows 0:80  ^T
#define W_2M      84480    // [80][96]    W_msg_h rows 80:160 ^T
#define W_3C      92160    // [160][128]  path weights (attn|msg) ^T, K-pad 102->128
#define W_ATOMO_T 112640   // [256][736]  W_atom_o^T, K-pad 722->736
#define W_TOTAL   301056

// ---------------- helpers ----------------
__device__ __forceinline__ __bf16 f2bf(float f) { return (__bf16)f; }   // hw v_cvt
__device__ __forceinline__ unsigned short bf2u(__bf16 b) {
  union { __bf16 b; unsigned short s; } v; v.b = b; return v.s;
}
__device__ __forceinline__ unsigned short f2bfu(float f) { return bf2u(f2bf(f)); }
__device__ __forceinline__ __bf16 u2bf(unsigned short s) {
  union { unsigned short s; __bf16 b; } v; v.s = s; return v.b;
}
__device__ __forceinline__ float bfu2f(unsigned short s) {
  union { unsigned u; float f; } v; v.u = ((unsigned)s) << 16; return v.f;
}
__device__ __forceinline__ v8f vzero8() {
  v8f z = {0.f,0.f,0.f,0.f,0.f,0.f,0.f,0.f}; return z;
}
// ISA 16-bit operand layout: lanes 0-15 hold K {0..7, 16..23}, lanes 16-31 hold K {8..15, 24..31}
__device__ __forceinline__ int frag_kloc(int lane, int t) {
  return ((t >> 3) << 4) + ((lane >> 4) << 3) + (t & 7);
}
// A/B fragment from row-major source where the 16 tile "rows" (M for A, N for B^T)
// each store K contiguously.  f32 source with K bound guard (converted via hw cvt):
__device__ __forceinline__ v16bf frag_f32(const float* base, int ld, int r0, int k0, int kmax) {
  int lane = threadIdx.x & 31;
  const float* row = base + (size_t)(r0 + (lane & 15)) * ld;
  v16bf v;
#pragma unroll
  for (int t = 0; t < 16; ++t) {
    int kk = k0 + frag_kloc(lane, t);
    float f = (kk < kmax) ? row[kk] : 0.0f;
    v[t] = f2bf(f);
  }
  return v;
}
// bf16 (ushort) source, fully padded: pure loads, no conversion
__device__ __forceinline__ v16bf frag_bf(const unsigned short* base, int ld, int r0, int k0) {
  int lane = threadIdx.x & 31;
  const unsigned short* row = base + (size_t)(r0 + (lane & 15)) * ld;
  v16bf v;
#pragma unroll
  for (int t = 0; t < 16; ++t) v[t] = u2bf(row[k0 + frag_kloc(lane, t)]);
  return v;
}
__device__ __forceinline__ v8f wmma_bf(v16bf a, v16bf b, v8f c) {
  return __builtin_amdgcn_wmma_f32_16x16x32_bf16(false, a, false, b, (short)0, c, false, false);
}

// ---------------- K1: weight transpose / pad / bf16 ----------------
__global__ void mol_prep_weights(const float* __restrict__ Wai, const float* __restrict__ Wah,
                                 const float* __restrict__ Wmh, const float* __restrict__ Wao,
                                 unsigned short* __restrict__ wt) {
  for (int idx = blockIdx.x * blockDim.x + threadIdx.x; idx < W_TOTAL;
       idx += gridDim.x * blockDim.x) {
    float val = 0.f;
    if (idx < W_1A)            { int l = idx;              int n = l / 96,  k = l % 96;  if (k < FA)  val = Wai[k * 640 + n]; }
    else if (idx < W_2A)       { int l = idx - W_1A;       int n = l / 96,  k = l % 96;  if (k < DK)  val = Wah[k * 80 + n]; }
    else if (idx < W_1M)       { int l = idx - W_2A;       int n = l / 96,  k = l % 96;  if (k < DK)  val = Wah[(80 + k) * 80 + n]; }
    else if (idx < W_2M)       { int l = idx - W_1M;       int n = l / 96,  k = l % 96;  if (k < DK)  val = Wmh[k * 80 + n]; }
    else if (idx < W_3C)       { int l = idx - W_2M;       int n = l / 96,  k = l % 96;  if (k < DK)  val = Wmh[(80 + k) * 80 + n]; }
    else if (idx < W_ATOMO_T)  { int l = idx - W_3C;       int n = l / 128, k = l % 128;
                                 if (k < FP) val = (n < 80) ? Wah[(160 + k) * 80 + n]
                                                            : Wmh[(160 + k) * 80 + (n - 80)]; }
    else                       { int l = idx - W_ATOMO_T;  int n = l / 736, k = l % 736; if (k < CIN) val = Wao[k * 256 + n]; }
    wt[idx] = f2bfu(val);
  }
}

// ---------------- K2: atom_h0 = atom_input @ W_atom_i  (head-major output) ----------------
__global__ void mol_atom_proj(const float* __restrict__ atom_in, const unsigned short* __restrict__ wt,
                              float* __restrict__ atom_h0, float* __restrict__ atom_h) {
  int r0 = blockIdx.x * 16;             // row in [2048] = b*64+n
  int h  = blockIdx.y;                  // head
  const unsigned short* W = wt + W_ATOMI_T;
  v8f c[5];
#pragma unroll
  for (int t = 0; t < 5; ++t) c[t] = vzero8();
  for (int k0 = 0; k0 < 96; k0 += 32) {
    v16bf a = frag_f32(atom_in, FA, r0, k0, FA);
#pragma unroll
    for (int t = 0; t < 5; ++t)
      c[t] = wmma_bf(a, frag_bf(W, 96, h * 80 + t * 16, k0), c[t]);
  }
  int lane = threadIdx.x & 31, cl = lane & 15, rb = (lane >> 4) << 3;
#pragma unroll
  for (int t = 0; t < 5; ++t)
#pragma unroll
    for (int r = 0; r < 8; ++r) {
      int R = r0 + rb + r, dk = t * 16 + cl;
      int bb = R >> 6, n = R & 63;
      size_t o = ((size_t)(h * NB + bb) * NN + n) * DK + dk;
      float v = c[t][r];
      atom_h0[o] = v; atom_h[o] = v;
    }
}

// ---------------- K3: P_attn / P_msg = path_input @ W3  (once, shared by heads & layers) ------
__global__ void mol_path_proj(const float* __restrict__ path_in, const unsigned short* __restrict__ wt,
                              unsigned short* __restrict__ P_attn, unsigned short* __restrict__ P_msg) {
  int r0 = blockIdx.x * 16;             // row in [131072] = (b*64+i)*64+j
  int sel = blockIdx.y;                 // 0 = attn, 1 = msg
  const unsigned short* W = wt + W_3C;
  unsigned short* dst = sel ? P_msg : P_attn;
  v8f c[5];
#pragma unroll
  for (int t = 0; t < 5; ++t) c[t] = vzero8();
  for (int k0 = 0; k0 < 128; k0 += 32) {
    v16bf a = frag_f32(path_in, FP, r0, k0, FP);
#pragma unroll
    for (int t = 0; t < 5; ++t)
      c[t] = wmma_bf(a, frag_bf(W, 128, sel * 80 + t * 16, k0), c[t]);
  }
  int lane = threadIdx.x & 31, cl = lane & 15, rb = (lane >> 4) << 3;
#pragma unroll
  for (int t = 0; t < 5; ++t)
#pragma unroll
    for (int r = 0; r < 8; ++r)
      dst[(size_t)(r0 + rb + r) * DK + t * 16 + cl] = f2bfu(c[t][r]);
}

// ---------------- K4: per-layer projections of atom_h ----------------
// sel 0..2 -> f32 outputs (Ai, Aj, AiM); sel 3 -> AjM stored TRANSPOSED bf16 [hb][80][64]
__global__ void mol_layer_proj(const float* __restrict__ atom_h, const unsigned short* __restrict__ wt,
                               float* __restrict__ Ai, float* __restrict__ Aj,
                               float* __restrict__ AiM, unsigned short* __restrict__ AjMt) {
  int r0 = blockIdx.x * 16;             // row in [16384] = hb*64+n
  int sel = blockIdx.y;
  const unsigned short* W;
  float* out = nullptr;
  switch (sel) {
    case 0:  W = wt + W_1A; out = Ai;  break;
    case 1:  W = wt + W_2A; out = Aj;  break;
    case 2:  W = wt + W_1M; out = AiM; break;
    default: W = wt + W_2M; break;
  }
  v8f c[5];
#pragma unroll
  for (int t = 0; t < 5; ++t) c[t] = vzero8();
  for (int k0 = 0; k0 < 96; k0 += 32) {
    v16bf a = frag_f32(atom_h, DK, r0, k0, DK);
#pragma unroll
    for (int t = 0; t < 5; ++t)
      c[t] = wmma_bf(a, frag_bf(W, 96, t * 16, k0), c[t]);
  }
  int lane = threadIdx.x & 31, cl = lane & 15, rb = (lane >> 4) << 3;
  if (sel < 3) {
#pragma unroll
    for (int t = 0; t < 5; ++t)
#pragma unroll
      for (int r = 0; r < 8; ++r)
        out[(size_t)(r0 + rb + r) * DK + t * 16 + cl] = c[t][r];
  } else {
    // transpose into [hb][dk][n] bf16 so mol_w2 B-fragments are contiguous loads
#pragma unroll
    for (int t = 0; t < 5; ++t)
#pragma unroll
      for (int r = 0; r < 8; ++r) {
        int R = r0 + rb + r;             // hb*64 + n
        int hb = R >> 6, n = R & 63, dk = t * 16 + cl;
        AjMt[((size_t)hb * DK + dk) * NN + n] = f2bfu(c[t][r]);
      }
  }
}

// ---------------- K5: score MLP + masked softmax (probs stored bf16) ----------------
__global__ void mol_scores(const float* __restrict__ Ai, const float* __restrict__ Aj,
                           const unsigned short* __restrict__ P_attn, const float* __restrict__ mask,
                           const float* __restrict__ b_attn_h, const float* __restrict__ W_attn_o,
                           const float* __restrict__ b_attn_o,
                           unsigned short* __restrict__ probs, float* __restrict__ rowsum) {
  int hbi = blockIdx.x;                 // hb*64 + i
  int hb = hbi >> 6, i = hbi & 63, b = hb & (NB - 1);
  int j = threadIdx.x;
  const float* AiR = Ai + (size_t)hbi * DK;
  const float* AjR = Aj + ((size_t)hb * NN + j) * DK;
  const unsigned short* P = P_attn + ((size_t)(b * NN + i) * NN + j) * DK;
  float m = mask[(size_t)(b * NN + i) * NN + j];
  float s = 0.f;
#pragma unroll 8
  for (int d = 0; d < DK; ++d) {
    float pre = AiR[d] + AjR[d] + bfu2f(P[d]) + b_attn_h[d];
    float hv = pre > 0.f ? pre : 0.2f * pre;      // leaky_relu(0.2)
    s += hv * W_attn_o[d];
  }
  s = (s + b_attn_o[0]) * m;
  __shared__ float red[64];
  red[j] = s; __syncthreads();
  for (int off = 32; off > 0; off >>= 1) { if (j < off) red[j] = fmaxf(red[j], red[j + off]); __syncthreads(); }
  float mx = red[0]; __syncthreads();
  float e = __expf(s - mx) * m;
  red[j] = e; __syncthreads();
  for (int off = 32; off > 0; off >>= 1) { if (j < off) red[j] += red[j + off]; __syncthreads(); }
  float S = red[0];
  probs[(size_t)hbi * NN + j] = f2bfu(e / (S + 1e-20f) * m);
  if (j == 0) rowsum[hbi] = S / (S + 1e-20f);
}

// ---------------- K6a: W2term = probs @ AjM  (all-bf16 fragments, zero convert VALU) --------
__global__ void mol_w2(const unsigned short* __restrict__ probs, const unsigned short* __restrict__ AjMt,
                       float* __restrict__ W2) {
  int hb = blockIdx.x;
  int wave = threadIdx.x >> 5;
  int r0 = wave * 16;
  const unsigned short* Pr = probs + (size_t)hb * NN * NN;  // [64 rows i][64 K=j]
  const unsigned short* Bm = AjMt + (size_t)hb * DK * NN;   // [80 rows dk][64 K=j]
  v8f c[5];
#pragma unroll
  for (int t = 0; t < 5; ++t) c[t] = vzero8();
  for (int k0 = 0; k0 < 64; k0 += 32) {
    v16bf a = frag_bf(Pr, NN, r0, k0);
#pragma unroll
    for (int t = 0; t < 5; ++t)
      c[t] = wmma_bf(a, frag_bf(Bm, NN, t * 16, k0), c[t]);
  }
  int lane = threadIdx.x & 31, cl = lane & 15, rb = (lane >> 4) << 3;
#pragma unroll
  for (int t = 0; t < 5; ++t)
#pragma unroll
    for (int r = 0; r < 8; ++r)
      W2[(size_t)hb * NN * DK + (size_t)(r0 + rb + r) * DK + t * 16 + cl] = c[t][r];
}

// ---------------- K6b: combine + Sum_j p*P_msg + bias + residual + relu ----------------
__global__ void mol_combine(const unsigned short* __restrict__ probs, const float* __restrict__ rowsum,
                            const float* __restrict__ AiM, const float* __restrict__ W2,
                            const unsigned short* __restrict__ P_msg, const float* __restrict__ b_msg_h,
                            const float* __restrict__ atom_h0, float* __restrict__ atom_h) {
  int hbi = blockIdx.x;
  int hb = hbi >> 6, i = hbi & 63, b = hb & (NB - 1);
  int d = threadIdx.x;
  if (d >= DK) return;
  const unsigned short* pr = probs + (size_t)hbi * NN;
  const unsigned short* pm = P_msg + (size_t)(b * NN + i) * NN * DK + d;
  float acc = 0.f;
#pragma unroll 8
  for (int jj = 0; jj < NN; ++jj) acc += bfu2f(pr[jj]) * bfu2f(pm[(size_t)jj * DK]);
  size_t o = (size_t)hbi * DK + d;
  float v = rowsum[hbi] * AiM[o] + W2[o] + acc + b_msg_h[d] + atom_h0[o];
  atom_h[o] = fmaxf(v, 0.f);
}

// ---------------- K7a: build concat [2048,736] bf16 ----------------
__global__ void mol_concat(const float* __restrict__ atom_in, const float* __restrict__ atom_h,
                           unsigned short* __restrict__ cc) {
  int total = 2048 * CPAD;
  for (int idx = blockIdx.x * blockDim.x + threadIdx.x; idx < total;
       idx += gridDim.x * blockDim.x) {
    int row = idx / CPAD, col = idx % CPAD;
    float v = 0.f;
    if (col < FA) v = atom_in[(size_t)row * FA + col];
    else if (col < CIN) {
      int c = col - FA, h = c / DK, dk = c % DK, b = row >> 6, n = row & 63;
      v = atom_h[((size_t)(h * NB + b) * NN + n) * DK + dk];
    }
    cc[idx] = f2bfu(v);
  }
}

// ---------------- K7b: out = relu(concat @ W_atom_o + b)  (WMMA) ----------------
__global__ void mol_final(const unsigned short* __restrict__ cc, const unsigned short* __restrict__ wt,
                          const float* __restrict__ bias, float* __restrict__ out) {
  int r0 = blockIdx.x * 16;
  int wave = threadIdx.x >> 5;
  int cb = wave * 64;
  const unsigned short* W = wt + W_ATOMO_T;
  v8f c[4];
#pragma unroll
  for (int t = 0; t < 4; ++t) c[t] = vzero8();
  for (int k0 = 0; k0 < CPAD; k0 += 32) {
    v16bf a = frag_bf(cc, CPAD, r0, k0);
#pragma unroll
    for (int t = 0; t < 4; ++t)
      c[t] = wmma_bf(a, frag_bf(W, CPAD, cb + t * 16, k0), c[t]);
  }
  int lane = threadIdx.x & 31, cl = lane & 15, rb = (lane >> 4) << 3;
#pragma unroll
  for (int t = 0; t < 4; ++t)
#pragma unroll
    for (int r = 0; r < 8; ++r) {
      int col = cb + t * 16 + cl, row = r0 + rb + r;
      out[(size_t)row * HID + col] = fmaxf(c[t][r] + bias[col], 0.f);
    }
}

// ---------------- launch ----------------
extern "C" void kernel_launch(void* const* d_in, const int* in_sizes, int n_in,
                              void* d_out, int out_size, void* d_ws, size_t ws_size,
                              hipStream_t stream) {
  (void)in_sizes; (void)n_in; (void)out_size; (void)ws_size;
  const float* atom_in = (const float*)d_in[0];
  const float* path_in = (const float*)d_in[1];
  const float* mask    = (const float*)d_in[2];
  const float* Wai     = (const float*)d_in[3];
  const float* Wah     = (const float*)d_in[4];
  const float* bah     = (const float*)d_in[5];
  const float* Wao_att = (const float*)d_in[6];
  const float* bao_att = (const float*)d_in[7];
  const float* Wmh     = (const float*)d_in[8];
  const float* bmh     = (const float*)d_in[9];
  const float* Wout    = (const float*)d_in[10];
  const float* bout    = (const float*)d_in[11];
  char* ws = (char*)d_ws;
  unsigned short* P_attn = (unsigned short*)(ws + OFF_P_ATTN);
  unsigned short* P_msg  = (unsigned short*)(ws + OFF_P_MSG);
  float* atom_h0 = (float*)(ws + OFF_ATOM_H0);
  float* atom_h  = (float*)(ws + OFF_ATOM_H);
  float* Ai      = (float*)(ws + OFF_AI);
  float* Aj      = (float*)(ws + OFF_AJ);
  float* AiM     = (float*)(ws + OFF_AIM);
  unsigned short* AjMt = (unsigned short*)(ws + OFF_AJMT);
  float* W2      = (float*)(ws + OFF_W2);
  unsigned short* probs = (unsigned short*)(ws + OFF_PROBS);
  float* rowsum  = (float*)(ws + OFF_ROWSUM);
  unsigned short* concat_bf = (unsigned short*)(ws + OFF_CONCAT);
  unsigned short* wt        = (unsigned short*)(ws + OFF_WTS);

  mol_prep_weights<<<(W_TOTAL + 255) / 256, 256, 0, stream>>>(Wai, Wah, Wmh, Wout, wt);
  mol_atom_proj<<<dim3(2048 / 16, HH), 32, 0, stream>>>(atom_in, wt, atom_h0, atom_h);
  mol_path_proj<<<dim3(131072 / 16, 2), 32, 0, stream>>>(path_in, wt, P_attn, P_msg);

  for (int layer = 0; layer < 2; ++layer) {
    mol_layer_proj<<<dim3(16384 / 16, 4), 32, 0, stream>>>(atom_h, wt, Ai, Aj, AiM, AjMt);
    mol_scores<<<16384, 64, 0, stream>>>(Ai, Aj, P_attn, mask, bah, Wao_att, bao_att,
                                         probs, rowsum);
    mol_w2<<<256, 128, 0, stream>>>(probs, AjMt, W2);
    mol_combine<<<16384, 96, 0, stream>>>(probs, rowsum, AiM, W2, P_msg, bmh, atom_h0, atom_h);
  }

  mol_concat<<<(2048 * CPAD + 255) / 256, 256, 0, stream>>>(atom_in, atom_h, concat_bf);
  mol_final<<<2048 / 16, 128, 0, stream>>>(concat_bf, wt, bout, (float*)d_out);
}
